// TensorizedPC_51359218925727
// MI455X (gfx1250) — matrix operand order repaired and checked by reading, not compiled, not run on hardware
//
#include <hip/hip_runtime.h>
#include <hip/hip_bf16.h>
#include <math.h>

typedef __attribute__((ext_vector_type(16))) _Float16 v16h;
typedef __attribute__((ext_vector_type(8)))  float    v8f;

#define ROWP 36     // padded row stride (floats) for [node][m][k] LDS tiles
#define NT   256    // threads per block (8 waves of 32)

#define LOG2E 1.4426950408889634f
#define LN2   0.6931471805599453f

// Raw hardware exp2/log2 (no libm denorm guards; args are safe by construction:
// exp2 input <= 0, log2 input >= min(W) = 0.01). Fallback must parse on host too.
__device__ __forceinline__ float hw_exp2(float x) {
#if __has_builtin(__builtin_amdgcn_exp2f)
  return __builtin_amdgcn_exp2f(x);
#else
  return exp2f(x);
#endif
}
__device__ __forceinline__ float hw_log2(float x) {
#if __has_builtin(__builtin_amdgcn_logf)
  return __builtin_amdgcn_logf(x);
#else
  return log2f(x);
#endif
}

// ---------------------------------------------------------------------------
// Weight prep: f32 weights[f][i][j] -> f16 wt[f][j][i]  (B-fragment friendly)
// ---------------------------------------------------------------------------
__global__ void k_prep_w(const float* __restrict__ w, _Float16* __restrict__ wt) {
  __shared__ float t[32 * 33];
  const int f = blockIdx.x;                      // 0..2046
  const float* wf = w + (size_t)f * 1024;
  for (int idx = threadIdx.x; idx < 1024; idx += NT) {
    int i = idx >> 5, j = idx & 31;
    t[i * 33 + j] = wf[idx];
  }
  __syncthreads();
  _Float16* o = wt + (size_t)f * 1024;
  for (int idx = threadIdx.x; idx < 1024; idx += NT) {
    int j = idx >> 5, i = idx & 31;
    o[idx] = (_Float16)t[i * 33 + j];           // out[f][j][i]
  }
}

// ---------------------------------------------------------------------------
// One tree-fold step for one node n (per wave), all values in LOG2 domain:
//   prod = cur[2n] + cur[2n+1]  (16 rows x 32 K, f32 in LDS)
//   nxt[n] = log2( exp2(prod - rowmax) @ W ) + rowmax
// via V_WMMA_F32_16X16X32_F16 (two N-halves).
// ---------------------------------------------------------------------------
__device__ __forceinline__ void node_step(const float* __restrict__ cur,
                                          float* __restrict__ nxt,
                                          const _Float16* __restrict__ wtf,
                                          int n) {
  const int lane = threadIdx.x & 31;
  const int m    = lane & 15;   // batch row for A fragment
  const int hi   = lane >> 4;   // lane half

  // A 16x32 layout (contiguous, matches B): lanes 0-15 hold K 0-15; lanes 16-31 K 16-31
  const int s0 = hi * 16;
  const int s1 = hi * 16 + 8;

  const float* c0 = cur + ((2 * n    ) * 16 + m) * ROWP;
  const float* c1 = cur + ((2 * n + 1) * 16 + m) * ROWP;

  float p[16];
#pragma unroll
  for (int q = 0; q < 2; ++q) {
    const int s = (q == 0) ? s0 : s1;
#pragma unroll
    for (int v = 0; v < 2; ++v) {
      float4 x0 = *(const float4*)(c0 + s + v * 4);
      float4 x1 = *(const float4*)(c1 + s + v * 4);
      const int b = q * 8 + v * 4;
      p[b + 0] = x0.x + x1.x;  p[b + 1] = x0.y + x1.y;
      p[b + 2] = x0.z + x1.z;  p[b + 3] = x0.w + x1.w;
    }
  }

  // row max over all 32 K (lane pair L, L^16 covers complementary K sets)
  float mx = p[0];
#pragma unroll
  for (int i = 1; i < 16; ++i) mx = fmaxf(mx, p[i]);
  mx = fmaxf(mx, __shfl_xor(mx, 16, 32));

  v16h a;
#pragma unroll
  for (int i = 0; i < 16; ++i) a[i] = (_Float16)hw_exp2(p[i] - mx);

  // B fragments: column j, contiguous 16 K starting at hi*16 (wt is [f][j][i])
  const v16h* bp = (const v16h*)wtf;
  v16h bf0 = bp[m * 2 + hi];           // columns j = 0..15
  v16h bf1 = bp[(16 + m) * 2 + hi];    // columns j = 16..31

  v8f z = {0.f, 0.f, 0.f, 0.f, 0.f, 0.f, 0.f, 0.f};
  v8f y0 = __builtin_amdgcn_wmma_f32_16x16x32_f16(false, a, false, bf0,
                                                  (short)0, z, false, false);
  v8f y1 = __builtin_amdgcn_wmma_f32_16x16x32_f16(false, a, false, bf1,
                                                  (short)0, z, false, false);

  // C layout: lane = N(+16), VGPR r = row r + 8*hi. Fetch each row's max by shuffle
  // (lane q<16 holds rowmax of row q).
  const int rbase = hi * 8;
#pragma unroll
  for (int r = 0; r < 8; ++r) {
    const float rm = __shfl(mx, rbase + r, 32);
    const int row = n * 16 + rbase + r;
    nxt[row * ROWP + m]      = hw_log2(y0[r]) + rm;   // j = m
    nxt[row * ROWP + 16 + m] = hw_log2(y1[r]) + rm;   // j = 16+m
  }
}

// Fold `levels` levels; F0 = output-node count at first level; gsub = subtree
// index (node base = gsub*(F0>>l)); Lg0 = first global level (weight offsets).
__device__ float* run_tree(float* bufA, float* bufB, const _Float16* __restrict__ wt,
                           int levels, int F0, int gsub, int Lg0) {
  float* cur = bufA;
  float* nxt = bufB;
  const int wid = threadIdx.x >> 5;
  for (int l = 0; l < levels; ++l) {
    const int Fo    = F0 >> l;
    const int Lg    = Lg0 + l;
    const int woff  = 2048 - (2048 >> Lg);   // sum of node counts of prior levels
    const int gbase = gsub * Fo;
    for (int n = wid; n < Fo; n += (NT / 32)) {
      const _Float16* wtf = wt + (size_t)(woff + gbase + n) * 1024;
      node_step(cur, nxt, wtf, n);
    }
    __syncthreads();
    float* t = cur; cur = nxt; nxt = t;
  }
  return cur;   // last-written buffer
}

// ---------------------------------------------------------------------------
// Stage 1: 16-batch-row x 32-leaf subtree per block; 5 levels in LDS.
// grid = (32 batch tiles, 64 subtrees); 108 KB LDS -> 2 blocks/WGP.
// Converts x from ln-domain to log2-domain on load.
// ---------------------------------------------------------------------------
__global__ void k_stage1(const float* __restrict__ x, const _Float16* __restrict__ wt,
                         float* __restrict__ s1out) {
  extern __shared__ float smem[];
  float* bufA = smem;                       // 32*16*ROWP floats
  float* bufB = smem + 32 * 16 * ROWP;      // 16*16*ROWP floats
  const int bt = blockIdx.x;                // batch tile
  const int g  = blockIdx.y;                // subtree (0..63)

  // Load x[bt*16 .. +15][32g .. +31][0..31] into bufA as [n][m][k] (row pad 36)
  const float* xb = x + (size_t)(bt * 16) * (2048 * 32) + (size_t)g * (32 * 32);
  for (int m = 0; m < 16; ++m) {
    const float* src = xb + (size_t)m * (2048 * 32);
    for (int idx = threadIdx.x; idx < 256; idx += NT) {   // 256 float4 = 1024 floats
      float4 v = *(const float4*)(src + idx * 4);
      v.x *= LOG2E; v.y *= LOG2E; v.z *= LOG2E; v.w *= LOG2E;
      const int k4 = idx * 4;
      const int n  = k4 >> 5;
      const int k  = k4 & 31;
      *(float4*)(bufA + (n * 16 + m) * ROWP + k) = v;
    }
  }
  __syncthreads();

  float* fin = run_tree(bufA, bufB, wt, /*levels=*/5, /*F0=*/16, /*gsub=*/g, /*Lg0=*/0);

  // store node-0 tile (log2 domain) to s1out[bt][g][m][k]
  float* dst = s1out + (size_t)(bt * 64 + g) * (16 * 32);
  for (int idx = threadIdx.x; idx < 512; idx += NT) {
    const int m = idx >> 5, k = idx & 31;
    dst[idx] = fin[m * ROWP + k];
  }
}

// ---------------------------------------------------------------------------
// Stage 2: remaining 6 levels (64 nodes -> 1) + mixture logsumexp.
// grid = 32 batch tiles; 216 KB LDS.
// ---------------------------------------------------------------------------
__global__ void k_stage2(const float* __restrict__ s1out, const _Float16* __restrict__ wt,
                         const float* __restrict__ mix_logw, float* __restrict__ out) {
  extern __shared__ float smem[];
  float* bufA = smem;                       // 64*16*ROWP floats
  float* bufB = smem + 64 * 16 * ROWP;      // 32*16*ROWP floats
  __shared__ float lw[32];
  const int bt = blockIdx.x;

  const float* src = s1out + (size_t)bt * (64 * 16 * 32);
  for (int idx = threadIdx.x; idx < (64 * 16 * 32) / 4; idx += NT) {
    float4 v = *(const float4*)(src + idx * 4);
    const int lin = idx * 4;
    const int n = lin >> 9;
    const int m = (lin >> 5) & 15;
    const int k = lin & 31;
    *(float4*)(bufA + (n * 16 + m) * ROWP + k) = v;
  }
  if (threadIdx.x == 0) {   // log_softmax(mix_logw) in ln-domain (tiny)
    float mx = mix_logw[0];
    for (int k = 1; k < 32; ++k) mx = fmaxf(mx, mix_logw[k]);
    float s = 0.f;
    for (int k = 0; k < 32; ++k) s += hw_exp2((mix_logw[k] - mx) * LOG2E);
    const float lse = hw_log2(s) * LN2 + mx;
    for (int k = 0; k < 32; ++k) lw[k] = mix_logw[k] - lse;
  }
  __syncthreads();

  float* fin = run_tree(bufA, bufB, wt, /*levels=*/6, /*F0=*/32, /*gsub=*/0, /*Lg0=*/5);

  if (threadIdx.x < 16) {
    const int m = threadIdx.x;
    float v[32];
    float mx = -3.0e38f;
    for (int k = 0; k < 32; ++k) {
      // root is log2-domain; reference flat = 2 * root_ln = 2*LN2*root_log2
      v[k] = (2.f * LN2) * fin[m * ROWP + k] + lw[k];
      mx = fmaxf(mx, v[k]);
    }
    float s = 0.f;
    for (int k = 0; k < 32; ++k) s += hw_exp2((v[k] - mx) * LOG2E);
    out[bt * 16 + m] = hw_log2(s) * LN2 + mx;
  }
}

// ---------------------------------------------------------------------------
extern "C" void kernel_launch(void* const* d_in, const int* in_sizes, int n_in,
                              void* d_out, int out_size, void* d_ws, size_t ws_size,
                              hipStream_t stream) {
  (void)in_sizes; (void)n_in; (void)out_size; (void)ws_size;
  const float* x   = (const float*)d_in[0];   // (512, 2048, 32) f32
  const float* w   = (const float*)d_in[1];   // (2047, 32, 32)  f32
  const float* mix = (const float*)d_in[3];   // (32,)           f32
  float* out = (float*)d_out;                 // (512,)          f32

  _Float16* wt  = (_Float16*)d_ws;                           // 2047*1024 f16 ~ 4.0 MB
  float*    s1o = (float*)((char*)d_ws + (size_t)(4 << 20)); // 32*64*16*32 f32 = 4 MB

  const size_t smem1 = (size_t)(32 * 16 * ROWP + 16 * 16 * ROWP) * sizeof(float); // 108 KB
  const size_t smem2 = (size_t)(64 * 16 * ROWP + 32 * 16 * ROWP) * sizeof(float); // 216 KB
  (void)hipFuncSetAttribute((const void*)k_stage1,
                            hipFuncAttributeMaxDynamicSharedMemorySize, (int)smem1);
  (void)hipFuncSetAttribute((const void*)k_stage2,
                            hipFuncAttributeMaxDynamicSharedMemorySize, (int)smem2);

  k_prep_w<<<2047, NT, 0, stream>>>(w, wt);
  k_stage1<<<dim3(32, 64), NT, smem1, stream>>>(x, wt, s1o);
  k_stage2<<<32, NT, smem2, stream>>>(s1o, wt, mix, out);
}
